// AttentionCNN_FE_SelfSupervise_12627203850695
// MI455X (gfx1250) — matrix operand order repaired and checked by reading, not compiled, not run on hardware
//
#include <hip/hip_runtime.h>

typedef __attribute__((ext_vector_type(16))) _Float16 v16h;
typedef __attribute__((ext_vector_type(8)))  _Float16 v8h;
typedef __attribute__((ext_vector_type(8)))  float    v8f;
typedef __attribute__((ext_vector_type(4)))  float    v4f;

#define BATCH   256
#define NPOS    121      // 11*11 patches
#define NCH     64
#define KDIM    192      // 8*8*3
#define APAD    200      // padded LDS row stride (f16) -> 400B rows, spreads banks

// ---------------------------------------------------------------------------
// Kernel 1: im2col GEMM via v_wmma_f32_16x16x32_f16.
// One block per image. 8 waves; wave = (mtile in 0..3, nhalf in 0..1).
// Two M-passes of 64 patch rows to keep static LDS <= 64KB.
// Fill is column-per-thread: koff computed once, patch origin incremental.
// ---------------------------------------------------------------------------
__global__ __launch_bounds__(256)
void conv_wmma_kernel(const int* __restrict__ rgb, const float* __restrict__ Wc,
                      const float* __restrict__ bc, float* __restrict__ vis_op)
{
    __shared__ __align__(32) _Float16 sB[NCH * APAD];   // sB[n][k] = Wc[k*64+n]
    __shared__ __align__(32) _Float16 sA[64 * APAD];    // 64 im2col rows per pass

    const int b    = blockIdx.x;
    const int tid  = threadIdx.x;
    const int lane = tid & 31;
    const int wave = tid >> 5;

    // Stage transposed weights (f32 -> f16), K contiguous per column.
    for (int i = tid; i < KDIM * NCH; i += 256) {
        const int k = i >> 6;          // 0..191
        const int n = i & 63;
        sB[n * APAD + k] = (_Float16)Wc[i];
    }

    // Per-thread K-column constants for the im2col fill (threads 0..191).
    // koff = offset of element k within the image, relative to patch origin:
    //   k = ky*24 + kx*3 + c  ->  koff = (ky*88 + kx)*3 + c
    int koff = 0;
    if (tid < KDIM) {
        const int ky  = tid / 24;
        const int rem = tid - ky * 24;
        const int kx  = rem / 3;
        const int c   = rem - kx * 3;
        koff = (ky * 88 + kx) * 3 + c;
    }
    const int imgBase = b * (88 * 88 * 3);

    const int mt    = wave & 3;                 // 16-row M tile within pass
    const int nh    = wave >> 2;                // 32-col N half
    const int mrow  = lane & 15;
    const int hi    = (lane >= 16) ? 1 : 0;
    const int koffA = hi ? 8  : 0;              // A: upper half-wave K+8 / K+24
    const int koffB = hi ? 8  : 0;              // B: upper half-wave K+8 / K+24
    const int ncol  = nh * 32 + (lane & 15);    // lane's column (tile 0)

    for (int pass = 0; pass < 2; ++pass) {
        const int pbase = pass * 64;
        __syncthreads();  // sB ready (pass 0); sA(pass 0) consumed (pass 1)

        // Column-per-thread im2col: thread k writes sA[r][k] for r = 0..63.
        if (tid < KDIM) {
            int oy = pbase / 11;                // pass0: 0 ; pass1: 5
            int ox = pbase - oy * 11;           // pass0: 0 ; pass1: 9
            int t  = oy * 88 + ox;              // patch origin in 8px units
            for (int r = 0; r < 64; ++r) {
                const int p = pbase + r;
                _Float16 v = (_Float16)0.0f;
                if (p < NPOS) {
                    const int idx = imgBase + 24 * t + koff;
                    v = (_Float16)((float)rgb[idx] * (1.0f / 255.0f));
                }
                sA[r * APAD + tid] = v;
                ++t;
                if (++ox == 11) { ox = 0; t += 77; }  // next patch row
            }
        }
        __syncthreads();

        v8f acc0 = {};
        v8f acc1 = {};
        const int arow = mt * 16 + mrow;

        #pragma unroll
        for (int ks = 0; ks < KDIM; ks += 32) {
            // A fragment: two contiguous 8xf16 chunks at K = ks+koffA and +16.
            const _Float16* ap = &sA[arow * APAD + ks + koffA];
            v8h alo = *(const v8h*)(ap);
            v8h ahi = *(const v8h*)(ap + 16);
            v16h av = __builtin_shufflevector(alo, ahi,
                        0,1,2,3,4,5,6,7,8,9,10,11,12,13,14,15);

            // B fragments: two 8xf16 chunks at K = ks+koffB and +16, columns ncol and ncol+16.
            const _Float16* bp0 = &sB[ncol * APAD + ks + koffB];
            v8h b0lo = *(const v8h*)(bp0);
            v8h b0hi = *(const v8h*)(bp0 + 16);
            v16h bv0 = __builtin_shufflevector(b0lo, b0hi,
                        0,1,2,3,4,5,6,7,8,9,10,11,12,13,14,15);

            const _Float16* bp1 = &sB[(ncol + 16) * APAD + ks + koffB];
            v8h b1lo = *(const v8h*)(bp1);
            v8h b1hi = *(const v8h*)(bp1 + 16);
            v16h bv1 = __builtin_shufflevector(b1lo, b1hi,
                        0,1,2,3,4,5,6,7,8,9,10,11,12,13,14,15);

            acc0 = __builtin_amdgcn_wmma_f32_16x16x32_f16(
                       false, av, false, bv0, (short)0, acc0, false, false);
            acc1 = __builtin_amdgcn_wmma_f32_16x16x32_f16(
                       false, av, false, bv1, (short)0, acc1, false, false);
        }

        // Epilogue: bias + ReLU, write fp32. D layout: VGPR r -> M = r + 8*hi.
        const int pr = pbase + mt * 16 + 8 * hi;
        const int n0 = nh * 32 + (lane & 15);
        const int n1 = n0 + 16;
        const float bias0 = bc[n0];
        const float bias1 = bc[n1];
        #pragma unroll
        for (int r = 0; r < 8; ++r) {
            const int p = pr + r;
            if (p < NPOS) {
                float v0 = acc0[r] + bias0; v0 = fmaxf(v0, 0.0f);
                float v1 = acc1[r] + bias1; v1 = fmaxf(v1, 0.0f);
                vis_op[(b * NPOS + p) * NCH + n0] = v0;
                vis_op[(b * NPOS + p) * NCH + n1] = v1;
            }
        }
    }
}

// ---------------------------------------------------------------------------
// Kernel 2: guidance. One block (128 thr) per image.
// counts -> histogram -> cumsum mask -> attn_i = keep_i/c_i / sum(keep_j/c_j)
// (softmax of -log(c) with -inf masking collapses to exactly this).
// ---------------------------------------------------------------------------
__global__ __launch_bounds__(128)
void guidance_kernel(const float* __restrict__ vis_op, float* __restrict__ guidance)
{
    __shared__ __align__(16) float sF[NPOS * NCH];
    __shared__ int   sC[NPOS];
    __shared__ int   sH[NPOS + 1];
    __shared__ int   sAllow[NPOS + 1];
    __shared__ float sW[NPOS];
    __shared__ float sSum;

    const int b   = blockIdx.x;
    const int tid = threadIdx.x;

    for (int i = tid; i < NPOS * NCH; i += 128)
        sF[i] = vis_op[b * NPOS * NCH + i];
    for (int i = tid; i < NPOS + 1; i += 128)
        sH[i] = 0;
    __syncthreads();

    if (tid < NPOS) {
        const v4f* fi = (const v4f*)&sF[tid * NCH];
        int cnt = 0;
        for (int j = 0; j < NPOS; ++j) {
            const v4f* fj = (const v4f*)&sF[j * NCH];
            bool eq = true;
            for (int q = 0; q < NCH / 4 && eq; ++q) {
                const v4f a = fi[q];
                const v4f c = fj[q];
                if (a.x != c.x || a.y != c.y || a.z != c.z || a.w != c.w)
                    eq = false;
            }
            cnt += eq ? 1 : 0;
        }
        sC[tid] = cnt;
        atomicAdd(&sH[cnt], 1);
    }
    __syncthreads();

    if (tid == 0) {
        int kmin = NPOS + 1;
        for (int i = 0; i < NPOS; ++i) kmin = min(kmin, sC[i]);
        const int Kthr = (int)(0.3 * NPOS);   // 36
        int cum = 0;
        for (int c = 0; c <= NPOS; ++c) {
            cum += sH[c];
            sAllow[c] = (cum <= Kthr) || (c == kmin);
        }
    }
    __syncthreads();

    float w = 0.0f;
    if (tid < NPOS) {
        const int c = sC[tid];
        if (sAllow[c]) w = 1.0f / (float)c;   // exp(-log c) = 1/c
        sW[tid] = w;
    }
    __syncthreads();

    if (tid == 0) {
        float s = 0.0f;
        for (int i = 0; i < NPOS; ++i) s += sW[i];
        sSum = s;
    }
    __syncthreads();

    if (tid < NPOS)
        guidance[b * NPOS + tid] = w / sSum;
}

// ---------------------------------------------------------------------------
extern "C" void kernel_launch(void* const* d_in, const int* in_sizes, int n_in,
                              void* d_out, int out_size, void* d_ws, size_t ws_size,
                              hipStream_t stream)
{
    const int*   rgb = (const int*)d_in[0];    // [256,88,88,3] int32
    const float* Wc  = (const float*)d_in[1];  // [8,8,3,64] f32
    const float* bc  = (const float*)d_in[2];  // [64] f32

    float* out  = (float*)d_out;
    float* vis  = out;                          // [256,121,64]
    float* guid = out + BATCH * NPOS * NCH;     // [256,1,11,11] == [256,121]

    conv_wmma_kernel<<<BATCH, 256, 0, stream>>>(rgb, Wc, bc, vis);
    guidance_kernel<<<BATCH, 128, 0, stream>>>(vis, guid);
}
